// UnstackedQwen3VLMoeSparseMoeBlock_60696477827344
// MI455X (gfx1250) — matrix operand order repaired and checked by reading, not compiled, not run on hardware
//
#include <hip/hip_runtime.h>
#include <math.h>

// Problem constants (from reference)
#define E_      32
#define TOPK_   8
#define H_      2048
#define I_      768
#define T_      2048   // B*S
#define TM_     32     // tokens per FFN block (2 row tiles of 16)
#define KC_     128    // K-chunk of hidden dim staged in LDS for FFN1
#define ICHUNK_ 192    // I columns per phase-1 pass (12 tiles, 3 per wave)
#define NCHUNK_ 1024   // H columns per phase-2 pass (64 tiles, 16 per wave)

typedef __attribute__((ext_vector_type(16))) _Float16 v16h;
typedef __attribute__((ext_vector_type(8)))  _Float16 v8h;
typedef __attribute__((ext_vector_type(8)))  float    v8f;

static __device__ __forceinline__ v8f wmma_f16(v16h a, v16h b, v8f c) {
  // D = A(16x32 f16) * B(32x16 f16) + C(16x16 f32)
  return __builtin_amdgcn_wmma_f32_16x16x32_f16(
      /*neg_a=*/false, a, /*neg_b=*/false, b,
      /*c_mod=*/(short)0, c, /*reuse_a=*/false, /*reuse_b=*/false);
}

// A-fragment (16x32, M x K, f16) from row-major LDS tile s[16][ld].
// Lanes 0-15 -> M=lane, K in {0..15}; lanes 16-31 -> M=lane-16,
// K in {16..31}. Two contiguous 16B runs per lane -> two ds_load_b128.
static __device__ __forceinline__ v16h load_a(const _Float16* __restrict__ s,
                                              int ld, int ks, int lane) {
  const _Float16* row = s + (lane & 15) * ld + ks + ((lane & 16) ? 16 : 0);
  v8h lo = *(const v8h*)(row);
  v8h hi = *(const v8h*)(row + 8);
  return __builtin_shufflevector(lo, hi, 0, 1, 2, 3, 4, 5, 6, 7,
                                 8, 9, 10, 11, 12, 13, 14, 15);
}

// B-fragment (32x16, K x N) for W stored row-major [N][K].
// Lanes 0-15 hold K=0..15 of column N=lane; lanes 16-31 hold K=16..31 of N=lane-16.
// fp32 overload: 4x b128 + 8x v_cvt_pk_f16_f32 (fallback path).
static __device__ __forceinline__ v16h load_b(const float* __restrict__ wbase,
                                              int ldk, int k0, int lane) {
  const float* p = wbase + (size_t)(lane & 15) * ldk + k0 + ((lane & 16) ? 16 : 0);
  __builtin_prefetch(p + 64, 0, 1);  // global_prefetch_b8: 256B ahead in K
  v16h b;
#pragma unroll
  for (int v = 0; v < 4; ++v) {
    float4 f = ((const float4*)p)[v];
    b[4 * v + 0] = (_Float16)f.x; b[4 * v + 1] = (_Float16)f.y;
    b[4 * v + 2] = (_Float16)f.z; b[4 * v + 3] = (_Float16)f.w;
  }
  return b;
}
// f16 overload: 2x b128, zero conversion VALU (fast path, pre-converted weights).
static __device__ __forceinline__ v16h load_b(const _Float16* __restrict__ wbase,
                                              int ldk, int k0, int lane) {
  const _Float16* p = wbase + (size_t)(lane & 15) * ldk + k0 + ((lane & 16) ? 16 : 0);
  __builtin_prefetch(p + 128, 0, 1);
  v8h lo = *(const v8h*)(p);
  v8h hi = *(const v8h*)(p + 8);
  return __builtin_shufflevector(lo, hi, 0, 1, 2, 3, 4, 5, 6, 7,
                                 8, 9, 10, 11, 12, 13, 14, 15);
}

// 16B gathered global->LDS async copy (GLOBAL_LOAD_ASYNC_TO_LDS_B128, ASYNCcnt).
// LDS destination address = low 32 bits of the generic pointer (LDS aperture).
static __device__ __forceinline__ void async_copy16(const void* gsrc, void* ldst) {
  unsigned lds_addr = (unsigned)(unsigned long long)ldst;
  asm volatile("global_load_async_to_lds_b128 %0, %1, off"
               :: "v"(lds_addr), "v"(gsrc) : "memory");
}
static __device__ __forceinline__ void wait_async0() {
  asm volatile("s_wait_asynccnt 0x0" ::: "memory");
}

// SiLU with fast reciprocal (v_rcp_f32 instead of the precise-div sequence;
// result is truncated to f16 anyway).
static __device__ __forceinline__ float silu_fast(float g) {
  return g * __builtin_amdgcn_rcpf(1.f + __expf(-g));
}

// ---------------------------------------------------------------------------
// fp32 -> f16 streaming converter (weights pre-pass for the fast path).
// ---------------------------------------------------------------------------
__global__ __launch_bounds__(256)
void cvt_f32_f16_kernel(const float* __restrict__ src,
                        _Float16* __restrict__ dst, long long n8) {
  long long i = (long long)blockIdx.x * blockDim.x + threadIdx.x;
  const long long stride = (long long)gridDim.x * blockDim.x;
  for (; i < n8; i += stride) {
    float4 a = ((const float4*)src)[2 * i];
    float4 b = ((const float4*)src)[2 * i + 1];
    v8h o;
    o[0] = (_Float16)a.x; o[1] = (_Float16)a.y;
    o[2] = (_Float16)a.z; o[3] = (_Float16)a.w;
    o[4] = (_Float16)b.x; o[5] = (_Float16)b.y;
    o[6] = (_Float16)b.z; o[7] = (_Float16)b.w;
    ((v8h*)dst)[i] = o;
  }
}

// ---------------------------------------------------------------------------
// Router: one wave32 per token. fp32 logits -> softmax -> deterministic top-8
// -> renormalized weights; builds per-expert token lists; f16 copy of x.
// ---------------------------------------------------------------------------
__global__ void moe_router_kernel(const float* __restrict__ x,
                                  const float* __restrict__ gate_w,
                                  _Float16* __restrict__ xb,
                                  float* __restrict__ R,
                                  int* __restrict__ counts,
                                  int* __restrict__ lists) {
  const int t = blockIdx.x;
  const int lane = threadIdx.x;  // lane == expert id
  const float* xrow = x + (size_t)t * H_;
  const float* grow = gate_w + (size_t)lane * H_;

  float acc = 0.f;
  for (int h = 0; h < H_; h += 4) {
    float4 xv = *(const float4*)(xrow + h);
    float4 gv = *(const float4*)(grow + h);
    acc = fmaf(xv.x, gv.x, acc); acc = fmaf(xv.y, gv.y, acc);
    acc = fmaf(xv.z, gv.z, acc); acc = fmaf(xv.w, gv.w, acc);
  }

  float mx = acc;
#pragma unroll
  for (int off = 16; off > 0; off >>= 1) mx = fmaxf(mx, __shfl_xor(mx, off, 32));
  float ex = __expf(acc - mx);
  float sum = ex;
#pragma unroll
  for (int off = 16; off > 0; off >>= 1) sum += __shfl_xor(sum, off, 32);
  const float prob = ex / sum;

  float val = prob;
  bool sel = false;
  float topsum = 0.f;
#pragma unroll
  for (int it = 0; it < TOPK_; ++it) {
    float v = val; int idx = lane;
#pragma unroll
    for (int off = 16; off > 0; off >>= 1) {
      float ov = __shfl_xor(v, off, 32);
      int   oi = __shfl_xor(idx, off, 32);
      if (ov > v || (ov == v && oi < idx)) { v = ov; idx = oi; }
    }
    topsum += v;
    if (lane == idx) { sel = true; val = -1.f; }
  }
  const float w = sel ? (prob / topsum) : 0.f;
  R[(size_t)t * E_ + lane] = w;
  if (sel) {
    int pos = atomicAdd(&counts[lane], 1);
    lists[(size_t)lane * T_ + pos] = t;
  }

  for (int h = lane * 4; h < H_; h += 32 * 4) {
    float4 xv = *(const float4*)(xrow + h);
    _Float16* d = xb + (size_t)t * H_ + h;
    d[0] = (_Float16)xv.x; d[1] = (_Float16)xv.y;
    d[2] = (_Float16)xv.z; d[3] = (_Float16)xv.w;
  }
}

// ---------------------------------------------------------------------------
// Fused expert FFN: block = 32-token tile of one expert, 8 waves.
// Wave (rw = wave>>2, cg = wave&3): rw picks the 16-row tile, cg a column group.
// Phase 1 (per 192-col I-chunk): g/u WMMA over K=H with async-staged X chunks,
// SiLU(g)*u -> LDS. Phase 2 (per 1024-col H-chunk): y = h @ Wd^T, weighted
// atomic scatter-add. WT = _Float16 (pre-converted weights) or float (inline cvt).
// ---------------------------------------------------------------------------
template <typename WT>
__global__ __launch_bounds__(256)
void moe_ffn_kernel(const _Float16* __restrict__ xb,
                    const WT* __restrict__ Wg,
                    const WT* __restrict__ Wu,
                    const WT* __restrict__ Wd,
                    const float* __restrict__ R,
                    const int* __restrict__ counts,
                    const int* __restrict__ lists,
                    float* __restrict__ out) {
  __shared__ __align__(16) _Float16 xs[TM_][KC_ + 8];  // 8.7 KB
  __shared__ __align__(16) _Float16 hb[TM_][I_ + 8];   // 48.5 KB
  __shared__ int   tok[TM_];
  __shared__ float wgt[TM_];

  const int e    = blockIdx.y;          // expert-major -> L2 reuse of weights
  const int n    = counts[e];
  const int base = blockIdx.x * TM_;
  if (base >= n) return;                // uniform exit: EXEC all-ones at WMMA

  const int tid   = threadIdx.x;
  const int wave  = tid >> 5;
  const int lane  = tid & 31;
  const int rw    = wave >> 2;          // row tile 0..1
  const int cg    = wave & 3;           // column group 0..3
  const int mbase = (lane >> 4) * 8;    // C/D layout: lanes>=16 -> M += 8
  const int ncol  = lane & 15;

  if (tid < TM_) {
    int idx = base + tid;
    bool valid = idx < n;
    int t = lists[(size_t)e * T_ + (valid ? idx : (n - 1))];
    tok[tid] = t;
    wgt[tid] = valid ? R[(size_t)t * E_ + e] : 0.f;
  }
  __syncthreads();

  const WT* wg_e = Wg + (size_t)e * I_ * H_;
  const WT* wu_e = Wu + (size_t)e * I_ * H_;
  v8f zero8 = {};

  // ---- Phase 1: gate & up projections ----
  for (int ic = 0; ic < I_; ic += ICHUNK_) {
    v8f gacc[3], uacc[3];
#pragma unroll
    for (int j = 0; j < 3; ++j) { gacc[j] = zero8; uacc[j] = zero8; }

    for (int kc = 0; kc < H_; kc += KC_) {
      // async-stage 32 gathered token rows x KC_ f16 (16B per transfer)
      for (int i = tid; i < TM_ * (KC_ / 8); i += 256) {
        int r  = i >> 4;               // KC_/8 == 16
        int cv = i & 15;
        async_copy16(xb + (size_t)tok[r] * H_ + kc + cv * 8, &xs[r][cv * 8]);
      }
      wait_async0();
      __syncthreads();

#pragma unroll 1
      for (int ks = 0; ks < KC_; ks += 32) {
        v16h a = load_a(&xs[rw * 16][0], KC_ + 8, ks, lane);
#pragma unroll
        for (int j = 0; j < 3; ++j) {
          int n0 = ic + (cg * 3 + j) * 16;
          v16h bg = load_b(wg_e + (size_t)n0 * H_, H_, kc + ks, lane);
          v16h bu = load_b(wu_e + (size_t)n0 * H_, H_, kc + ks, lane);
          gacc[j] = wmma_f16(a, bg, gacc[j]);
          uacc[j] = wmma_f16(a, bu, uacc[j]);
        }
      }
      __syncthreads();  // xs overwritten next chunk
    }

    // SiLU(g) * u -> LDS h staging for this I-chunk
#pragma unroll
    for (int j = 0; j < 3; ++j) {
      int n0 = ic + (cg * 3 + j) * 16;
#pragma unroll
      for (int p = 0; p < 8; ++p) {
        float g = gacc[j][p];
        float u = uacc[j][p];
        hb[rw * 16 + mbase + p][n0 + ncol] = (_Float16)(silu_fast(g) * u);
      }
    }
  }
  __syncthreads();

  // ---- Phase 2: down projection ----
  const WT* wd_e = Wd + (size_t)e * H_ * I_;
#pragma unroll 1
  for (int nc = 0; nc < H_; nc += NCHUNK_) {
    v8f yacc[16];
#pragma unroll
    for (int j = 0; j < 16; ++j) yacc[j] = zero8;

#pragma unroll 1
    for (int ks = 0; ks < I_; ks += 32) {
      v16h a = load_a(&hb[rw * 16][0], I_ + 8, ks, lane);
#pragma unroll
      for (int j = 0; j < 16; ++j) {
        int n0 = nc + (cg * 16 + j) * 16;
        v16h bd = load_b(wd_e + (size_t)n0 * I_, I_, ks, lane);
        yacc[j] = wmma_f16(a, bd, yacc[j]);
      }
    }

    // routing-weighted scatter-add
#pragma unroll
    for (int j = 0; j < 16; ++j) {
      int n0 = nc + (cg * 16 + j) * 16 + ncol;
#pragma unroll
      for (int p = 0; p < 8; ++p) {
        int m = rw * 16 + mbase + p;
        atomicAdd(&out[(size_t)tok[m] * H_ + n0], yacc[j][p] * wgt[m]);
      }
    }
  }
}

// ---------------------------------------------------------------------------
// Host launch. Inputs: hidden_states[T,H] f32, gate_w[E,H] f32,
// Wg[E,I,H] f32, Wu[E,I,H] f32, Wd[E,H,I] f32. Output: [T,H] f32.
// Workspace: base ~8.9 MB (xb/R/lists/counts). If ws_size additionally fits
// 3x 100.7 MB of f16 weights, take the fast path: one-shot fp32->f16 weight
// conversion (~0.9 GB streamed once) then a conversion-free WMMA hot loop.
// ---------------------------------------------------------------------------
extern "C" void kernel_launch(void* const* d_in, const int* in_sizes, int n_in,
                              void* d_out, int out_size, void* d_ws, size_t ws_size,
                              hipStream_t stream) {
  (void)in_sizes; (void)n_in;
  const float* x      = (const float*)d_in[0];
  const float* gate_w = (const float*)d_in[1];
  const float* Wg     = (const float*)d_in[2];
  const float* Wu     = (const float*)d_in[3];
  const float* Wd     = (const float*)d_in[4];
  float* out = (float*)d_out;

  char* ws = (char*)d_ws;
  size_t off = 0;
  _Float16* xb = (_Float16*)(ws + off); off += (size_t)T_ * H_ * sizeof(_Float16);
  float*    R  = (float*)(ws + off);    off += (size_t)T_ * E_ * sizeof(float);
  int*   lists = (int*)(ws + off);      off += (size_t)E_ * T_ * sizeof(int);
  int*  counts = (int*)(ws + off);      off += 256;

  const size_t welems = (size_t)E_ * I_ * H_;        // per weight tensor
  const size_t wbytes = welems * sizeof(_Float16);   // 100.7 MB
  const bool fast = ws_size >= off + 3 * wbytes;

  hipMemsetAsync(out, 0, (size_t)out_size * sizeof(float), stream);
  hipMemsetAsync(counts, 0, E_ * sizeof(int), stream);

  moe_router_kernel<<<dim3(T_), dim3(32), 0, stream>>>(x, gate_w, xb, R, counts, lists);

  if (fast) {
    _Float16* wgH = (_Float16*)(ws + off);
    _Float16* wuH = (_Float16*)(ws + off + wbytes);
    _Float16* wdH = (_Float16*)(ws + off + 2 * wbytes);
    const long long n8 = (long long)(welems / 8);
    cvt_f32_f16_kernel<<<dim3(4096), dim3(256), 0, stream>>>(Wg, wgH, n8);
    cvt_f32_f16_kernel<<<dim3(4096), dim3(256), 0, stream>>>(Wu, wuH, n8);
    cvt_f32_f16_kernel<<<dim3(4096), dim3(256), 0, stream>>>(Wd, wdH, n8);
    moe_ffn_kernel<_Float16><<<dim3(T_ / TM_, E_), dim3(256), 0, stream>>>(
        xb, wgH, wuH, wdH, R, counts, lists, out);
  } else {
    moe_ffn_kernel<float><<<dim3(T_ / TM_, E_), dim3(256), 0, stream>>>(
        xb, Wg, Wu, Wd, R, counts, lists, out);
  }
}
